// HybridTritonAdaptivePiecewiseConv2d_22351009808403
// MI455X (gfx1250) — matrix operand order repaired and checked by reading, not compiled, not run on hardware
//
#include <hip/hip_runtime.h>
#include <hip/hip_bf16.h>

typedef __attribute__((ext_vector_type(16))) _Float16 v16h;
typedef __attribute__((ext_vector_type(8)))  _Float16 v8h;
typedef __attribute__((ext_vector_type(4)))  _Float16 v4h;
typedef __attribute__((ext_vector_type(8)))  float    v8f;

#define B_    8
#define CIN_  16
#define H_    32
#define W_    32
#define COUT_ 32
#define KK_   144      // CIN*KH*KW
#define KP_   432      // KK_*3 (3 hat-basis functions per tap)
#define NCHUNK_ 14     // K' padded to 448
#define BM_   64       // output rows per block (= 2 image rows)
#define LDP_  456      // padded LDS row stride in halfs (912B = 57*16B)
#define XLD_  36       // x-tile column stride (34 used + pad)

// Fused adaptive-piecewise conv: hat-basis expansion (VALU -> LDS f16)
// then GEMM Phi[64,448] x W[32,448]^T via v_wmma_f32_16x16x32_f16.
__global__ __launch_bounds__(128)
void apw_conv_wmma_kernel(const float* __restrict__ x,
                          const float* __restrict__ positions,
                          const float* __restrict__ values,
                          float* __restrict__ out)
{
    __shared__ __align__(16) _Float16 sW[COUT_ * LDP_];    // 29184 B
    __shared__ __align__(16) _Float16 sPhi[BM_ * LDP_];    // 58368 B
    __shared__ __align__(16) float sX[CIN_ * 4 * XLD_];    //  9216 B (halo'd x tile)
    __shared__ float sPos[KK_ * 3];                        //  1728 B
    __shared__ float sInv[KK_ * 2];                        //  1152 B

    const int t    = threadIdx.x;
    const int row0 = blockIdx.x * BM_;     // 64 rows, one batch image
    const int b    = row0 >> 10;
    const int oh0  = (row0 & 1023) >> 5;   // first of 2 image rows (even)

    // ---- stage W: 4 threads per cout row, 27 x float4 each (no guards) ----
    {
        const int o   = t >> 2;            // 0..31
        const int seg = t & 3;             // 0..3, 108 floats each
        const float4* __restrict__ src = (const float4*)(values + o * KP_ + seg * 108);
        _Float16* dst = &sW[o * LDP_ + seg * 108];
        for (int j = 0; j < 27; ++j) {
            const float4 v = src[j];
            v4h h = { (_Float16)v.x, (_Float16)v.y, (_Float16)v.z, (_Float16)v.w };
            *(v4h*)(dst + 4 * j) = h;      // ds_store_b64, 8B aligned
        }
    }
    // ---- zero-pad W columns [432,456): 96 threads x 8 halfs ----
    if (t < 96) {
        const int o = t / 3, part = t - 3 * o;
        _Float16* dst = &sW[o * LDP_ + KP_ + part * 8];
        v4h z = {};
        *(v4h*)dst = z; *(v4h*)(dst + 4) = z;
    }
    // ---- zero-pad Phi columns [432,456) ----
    for (int idx = t; idx < BM_ * 3; idx += 128) {
        const int r = idx / 3, part = idx - 3 * r;
        _Float16* dst = &sPhi[r * LDP_ + KP_ + part * 8];
        v4h z = {};
        *(v4h*)dst = z; *(v4h*)(dst + 4) = z;
    }
    // ---- stage knots (shared across cout) + reciprocal spacings ----
    for (int k = t; k < KK_; k += 128) {
        const float p0 = positions[k * 3 + 0];
        const float p1 = positions[k * 3 + 1];
        const float p2 = positions[k * 3 + 2];
        sPos[k * 3 + 0] = p0; sPos[k * 3 + 1] = p1; sPos[k * 3 + 2] = p2;
        const float d01 = p1 - p0, d12 = p2 - p1;
        sInv[k * 2 + 0] = (d01 != 0.0f) ? 1.0f / d01 : 0.0f; // interval empty if 0
        sInv[k * 2 + 1] = (d12 != 0.0f) ? 1.0f / d12 : 0.0f;
    }
    // ---- stage x tile with halo, zero-padded: [cin][4 rows][36 cols] ----
    {
        const int pair = t >> 1;           // cin*4 + rrow (0..63)
        const int sub  = t & 1;            // half of the 36 columns
        const int cin  = pair >> 2;
        const int rrow = pair & 3;
        const int ih   = oh0 - 1 + rrow;
        const bool ihok = (unsigned)ih < (unsigned)H_;
        const float* __restrict__ srcX = x + ((b * CIN_ + cin) * H_ + ih) * W_;
        float* dstX = &sX[pair * XLD_ + sub * 18];
        #pragma unroll
        for (int j = 0; j < 18; ++j) {
            const int iw = sub * 18 + j - 1;
            float v = 0.0f;
            if (ihok && (unsigned)iw < (unsigned)W_) v = srcX[iw];
            dstX[j] = v;
        }
    }

    __syncthreads();   // sX / sPos / sInv ready

    // ---- Phi fill: thread -> (row m_local, half of cin range); no bounds checks ----
    const int m_local = t & 63;
    const int half    = t >> 6;            // cin in [half*8, half*8+8)
    const int hw  = (row0 + m_local) & 1023;
    const int ohl = (hw >> 5) - oh0;       // 0 or 1
    const int ow  = hw & 31;

    for (int c8 = 0; c8 < 8; ++c8) {
        const int cin = half * 8 + c8;
        const float* __restrict__ xrow = &sX[(cin * 4 + ohl) * XLD_ + ow];
        #pragma unroll
        for (int r = 0; r < 9; ++r) {
            const int kh = r / 3, kw = r % 3;          // compile-time after unroll
            const float xv = xrow[kh * XLD_ + kw];     // halo/zero-pad baked in
            const int k = cin * 9 + r;
            const float p0 = sPos[k * 3 + 0];
            const float p1 = sPos[k * 3 + 1];
            const float p2 = sPos[k * 3 + 2];
            const float i01 = sInv[k * 2 + 0];
            const float i12 = sInv[k * 2 + 1];
            // hat basis functions (exact reference boundary semantics)
            const float phi0 = (xv <= p0) ? 1.0f
                             : ((xv < p1) ? (p1 - xv) * i01 : 0.0f);
            float phi1;
            if (xv > p0 && xv < p1)        phi1 = (xv - p0) * i01;
            else if (xv >= p1 && xv < p2)  phi1 = (p2 - xv) * i12;
            else                           phi1 = 0.0f;
            const float phi2 = (xv >= p2) ? 1.0f
                             : ((xv >= p1) ? (xv - p1) * i12 : 0.0f);
            _Float16* d = &sPhi[m_local * LDP_ + k * 3];
            d[0] = (_Float16)phi0;
            d[1] = (_Float16)phi1;
            d[2] = (_Float16)phi2;
        }
    }

    __syncthreads();   // Phi + W tiles ready

    // ---- pure WMMA sweep: wave w owns rows 16w..16w+15, all 32 couts ----
    const int wave = t >> 5, lane = t & 31;
    const int mrow = wave * 16 + (lane & 15);
    const int koff = (lane >> 4) * 16;     // A layout: 16 consecutive K per lane
    const int nlo  = lane & 15;
    const int kr   = (lane >> 4) * 16;     // B layout: 16 consecutive K per lane

    const _Float16* __restrict__ aRow  = &sPhi[mrow * LDP_];
    const _Float16* __restrict__ b0Row = &sW[(nlo     ) * LDP_];
    const _Float16* __restrict__ b1Row = &sW[(nlo + 16) * LDP_];

    v8f acc0 = {}; v8f acc1 = {};
    #pragma unroll
    for (int c = 0; c < NCHUNK_; ++c) {
        const int kb = c * 32;
        union { v16h v; v8h h[2]; } A, B0, B1;
        A.h[0]  = *(const v8h*)(aRow  + kb + koff);
        A.h[1]  = *(const v8h*)(aRow  + kb + koff + 8);
        B0.h[0] = *(const v8h*)(b0Row + kb + kr);
        B0.h[1] = *(const v8h*)(b0Row + kb + kr + 8);
        B1.h[0] = *(const v8h*)(b1Row + kb + kr);
        B1.h[1] = *(const v8h*)(b1Row + kb + kr + 8);
        acc0 = __builtin_amdgcn_wmma_f32_16x16x32_f16(false, A.v, false, B0.v,
                                                      (short)0, acc0, false, false);
        acc1 = __builtin_amdgcn_wmma_f32_16x16x32_f16(false, A.v, false, B1.v,
                                                      (short)0, acc1, false, false);
    }

    // ---- write D (16x16 f32 C/D layout: vgpr r -> M=r+8*(lane>=16), N=lane&15)
    const int n     = lane & 15;
    const int mbase = row0 + wave * 16 + ((lane >> 4) * 8);
    #pragma unroll
    for (int r = 0; r < 8; ++r) {
        const int phw = (mbase + r) & 1023;
        out[((b * COUT_ + n     ) << 10) + phw] = acc0[r];
        out[((b * COUT_ + n + 16) << 10) + phw] = acc1[r];
    }
}

extern "C" void kernel_launch(void* const* d_in, const int* in_sizes, int n_in,
                              void* d_out, int out_size, void* d_ws, size_t ws_size,
                              hipStream_t stream) {
    (void)in_sizes; (void)n_in; (void)out_size; (void)d_ws; (void)ws_size;
    const float* x         = (const float*)d_in[0];
    const float* positions = (const float*)d_in[1];
    const float* values    = (const float*)d_in[2];
    float* out             = (float*)d_out;

    const int M = B_ * H_ * W_;            // 8192
    dim3 grid(M / BM_);                    // 128 blocks
    dim3 block(128);                       // 4 wave32
    apw_conv_wmma_kernel<<<grid, block, 0, stream>>>(x, positions, values, out);
}
